// AdvancedGNNPredictorV2_53420803228016
// MI455X (gfx1250) — matrix operand order repaired and checked by reading, not compiled, not run on hardware
//
#include <hip/hip_runtime.h>

typedef __attribute__((ext_vector_type(16))) __bf16 v16bf;
typedef __attribute__((ext_vector_type(8)))  float  v8f;

#define SLOPE 0.2f
#define EPSBN 1e-5f

static __device__ __forceinline__ unsigned short f2bf(float f) {
  unsigned int u = __float_as_uint(f);
  unsigned int r = u + 0x7fffu + ((u >> 16) & 1u);   // round-to-nearest-even
  return (unsigned short)(r >> 16);
}

static __device__ __forceinline__ void atomicMaxF(float* addr, float val) {
  if (val >= 0.0f) atomicMax((int*)addr, __float_as_int(val));
  else             atomicMin((unsigned int*)addr, __float_as_uint(val));
}

// ---------------------------------------------------------------- elementwise
__global__ __launch_bounds__(256) void cvt_bf16_kernel(const float* __restrict__ in,
                                                       unsigned short* __restrict__ out, int n) {
  int i = blockIdx.x * blockDim.x + threadIdx.x;
  if (i < n) out[i] = f2bf(in[i]);
}

// transpose-convert: in[K,Nc] f32 row-major -> out[Nc,K] bf16
__global__ __launch_bounds__(256) void cvt_bf16_t_kernel(const float* __restrict__ in,
                                                         unsigned short* __restrict__ out,
                                                         int K, int Nc) {
  int i = blockIdx.x * blockDim.x + threadIdx.x;
  if (i >= K * Nc) return;
  int k = i / Nc, c = i - k * Nc;
  out[(size_t)c * K + k] = f2bf(in[i]);
}

__global__ __launch_bounds__(256) void add_vec_kernel(const float* __restrict__ a,
                                                      const float* __restrict__ b,
                                                      float* __restrict__ o, int n) {
  int i = blockIdx.x * blockDim.x + threadIdx.x;
  if (i < n) o[i] = a[i] + b[i];
}

__global__ __launch_bounds__(256) void fill_f32_kernel(float* __restrict__ p, float v, int n) {
  int i = blockIdx.x * blockDim.x + threadIdx.x;
  if (i < n) p[i] = v;
}

// ---------------------------------------------------------------- WMMA GEMM
// C[M,Nc] = A_bf16[M,K] * Bt_bf16[Nc,K]^T (+ bias) (optional relu)
// block: 256 threads = 8 waves -> 4 (M) x 2 (N) grid of 16x16 wave tiles = 64x32
// A tile staged in LDS via double-buffered global_load_async_to_lds_b128.
// B fragments loaded straight from global (weights resident in L2/WGP$).
// requires K % 32 == 0, Nc % 32 == 0 (grid.x = Nc/32)
#define BM 64
#define BN 32
#define BK 32

__global__ __launch_bounds__(256) void gemm_bf16_wmma(
    const unsigned short* __restrict__ A,    // [M,K] bf16 row-major
    const unsigned short* __restrict__ Bt,   // [Nc,K] bf16 (pre-transposed weights)
    const float* __restrict__ bias,
    float* __restrict__ C,
    int M, int K, int Nc, int do_relu)
{
  __shared__ __align__(16) unsigned short As[2][BM][BK];   // 2 x 4KB, 64B row stride

  const int t       = threadIdx.x;
  const int lane    = t & 31;
  const int wid     = t >> 5;
  const int waveM   = wid & 3;
  const int waveN   = wid >> 2;
  const int m0      = blockIdx.y * BM;
  const int n0      = blockIdx.x * BN;
  const int halfSel = (lane >> 4) & 1;    // lanes 16..31: other K/M half
  const int r16     = lane & 15;

  // async tile copy: one 16B packet per thread (256 x 16B = 64x32 bf16 tile)
  const int arow = t >> 2;
  const int acol = (t & 3) << 3;
  const int agr  = m0 + arow;

  const int bcol = n0 + waveN * 16 + r16;
  const unsigned short* bbase = Bt + (size_t)bcol * K + halfSel * 16;
  __builtin_prefetch(bbase, 0, 3);        // global_prefetch_b8 (weights -> caches)

  v8f acc = {};
  const int nk = K >> 5;

  // prologue: stage K-tile 0 into As[0]
  if (agr < M) {
    unsigned int ldsa = (unsigned int)(unsigned long long)&As[0][arow][acol];
    unsigned long long ga = (unsigned long long)(A + (size_t)agr * K + acol);
    asm volatile("global_load_async_to_lds_b128 %0, %1, off"
                 :: "v"(ldsa), "v"(ga) : "memory");
  }

  for (int kt = 0; kt < nk; ++kt) {
    asm volatile("s_wait_asynccnt 0" ::: "memory");   // my async writes done
    __syncthreads();                                   // everyone's writes/reads done
    if ((kt + 1 < nk) && (agr < M)) {                  // kick next tile into other buffer
      unsigned int ldsa = (unsigned int)(unsigned long long)&As[(kt + 1) & 1][arow][acol];
      unsigned long long ga =
          (unsigned long long)(A + (size_t)agr * K + ((kt + 1) << 5) + acol);
      asm volatile("global_load_async_to_lds_b128 %0, %1, off"
                   :: "v"(ldsa), "v"(ga) : "memory");
    }

    // A fragment: two contiguous 16B runs from LDS (ds_load_b128 x2)
    union { uint4 q[2]; v16bf v; } af, bf;
    const unsigned short* ap = &As[kt & 1][waveM * 16 + r16][halfSel * 16];
    af.q[0] = *(const uint4*)(ap);
    af.q[1] = *(const uint4*)(ap + 8);
    // B fragment: 16 contiguous bf16 from transposed weights (global_load_b128 x2)
    const unsigned short* bp = bbase + (kt << 5);
    bf.q[0] = *(const uint4*)(bp);
    bf.q[1] = *(const uint4*)(bp + 8);

    acc = __builtin_amdgcn_wmma_f32_16x16x32_bf16(false, af.v, false, bf.v,
                                                  (short)0, acc, false, false);
  }

  // C/D layout: VGPR v -> M = v (+8 for lanes 16..31), N = lane%16
#pragma unroll
  for (int v = 0; v < 8; ++v) {
    int gr = m0 + waveM * 16 + v + halfSel * 8;
    int gc = n0 + waveN * 16 + r16;
    if (gr < M) {
      float val = acc[v];
      if (bias) val += bias[gc];
      if (do_relu) val = fmaxf(val, 0.0f);
      C[(size_t)gr * Nc + gc] = val;
    }
  }
}

// ---------------------------------------------------------------- attention
__global__ __launch_bounds__(256) void attn_coef_kernel(
    const float* __restrict__ lin, const float* __restrict__ asrc,
    const float* __restrict__ adst, float* __restrict__ es,
    float* __restrict__ ed, int total /* N*4 */)
{
  int i = blockIdx.x * blockDim.x + threadIdx.x;
  if (i >= total) return;
  int n = i >> 2, h = i & 3;
  const float* hp = lin + (size_t)n * 256 + h * 64;
  const float* ap = asrc + h * 64;
  const float* dp = adst + h * 64;
  float s = 0.0f, d = 0.0f;
#pragma unroll 8
  for (int c = 0; c < 64; ++c) { float v = hp[c]; s += v * ap[c]; d += v * dp[c]; }
  es[i] = s; ed[i] = d;
}

__global__ __launch_bounds__(256) void edge_logit_max_kernel(
    const int* __restrict__ src, const int* __restrict__ dst,
    const float* __restrict__ es, const float* __restrict__ ed,
    float* __restrict__ eraw, float* __restrict__ emax, int total /* E*4 */)
{
  int i = blockIdx.x * blockDim.x + threadIdx.x;
  if (i >= total) return;
  int e = i >> 2, h = i & 3;
  int s = src[e], d = dst[e];
  float v = es[s * 4 + h] + ed[d * 4 + h];
  v = (v > 0.0f) ? v : SLOPE * v;
  eraw[i] = v;
  atomicMaxF(&emax[d * 4 + h], v);
}

__global__ __launch_bounds__(256) void edge_exp_den_kernel(
    const int* __restrict__ dst, const float* __restrict__ emax,
    float* __restrict__ ex, float* __restrict__ den, int total /* E*4 */)
{
  int i = blockIdx.x * blockDim.x + threadIdx.x;
  if (i >= total) return;
  int e = i >> 2, h = i & 3;
  int d = dst[e];
  float m = emax[d * 4 + h];
  m = (m > -1e29f) ? m : 0.0f;   // empty segments -> 0 (matches reference)
  float v = __expf(ex[i] - m);
  ex[i] = v;
  atomicAdd(&den[d * 4 + h], v);
}

__global__ __launch_bounds__(256) void edge_scatter_kernel(
    const int* __restrict__ src, const int* __restrict__ dst,
    const float* __restrict__ lin, const float* __restrict__ ex,
    const float* __restrict__ den, float* __restrict__ agg, int total /* E*64 */)
{
  int i = blockIdx.x * blockDim.x + threadIdx.x;
  if (i >= total) return;
  int e  = i >> 6;
  int cb = (i & 63) << 2;
  int h  = cb >> 6;
  int s = src[e], d = dst[e];
  float alpha = ex[e * 4 + h] / (den[d * 4 + h] + 1e-16f);
  const float4 hv = *(const float4*)(lin + (size_t)s * 256 + cb);
  float* ap = agg + (size_t)d * 256 + cb;
  atomicAdd(ap + 0, hv.x * alpha);
  atomicAdd(ap + 1, hv.y * alpha);
  atomicAdd(ap + 2, hv.z * alpha);
  atomicAdd(ap + 3, hv.w * alpha);
}

// ---------------------------------------------------------------- batchnorm
__global__ __launch_bounds__(256) void bn_stats_kernel(
    const float* __restrict__ xin, float* __restrict__ sum,
    float* __restrict__ sumsq, int N)
{
  int c  = threadIdx.x;
  int rpb = (N + gridDim.x - 1) / gridDim.x;
  int r0 = blockIdx.x * rpb;
  int r1 = min(N, r0 + rpb);
  float s = 0.0f, q = 0.0f;
  for (int r = r0; r < r1; ++r) {
    float v = xin[(size_t)r * 256 + c];
    s += v; q += v * v;
  }
  atomicAdd(&sum[c], s);
  atomicAdd(&sumsq[c], q);
}

__global__ __launch_bounds__(256) void bn_finalize_kernel(
    const float* __restrict__ sum, const float* __restrict__ sumsq,
    float* __restrict__ mean, float* __restrict__ rstd, int N)
{
  int c = threadIdx.x;
  float m = sum[c] / (float)N;
  float v = sumsq[c] / (float)N - m * m;
  mean[c] = m;
  rstd[c] = rsqrtf(v + EPSBN);
}

__global__ __launch_bounds__(256) void bn_apply_kernel(
    const float* __restrict__ xin, const float* __restrict__ mean,
    const float* __restrict__ rstd, const float* __restrict__ g,
    const float* __restrict__ b, float* __restrict__ outf,
    unsigned short* __restrict__ outb, int total /* N*256 */)
{
  int i = blockIdx.x * blockDim.x + threadIdx.x;
  if (i >= total) return;
  int c = i & 255;
  float v = (xin[i] - mean[c]) * rstd[c] * g[c] + b[c];
  v = fmaxf(v, 0.0f);
  if (outf) outf[i] = v;
  if (outb) outb[i] = f2bf(v);
}

// ---------------------------------------------------------------- pooling/MLP
__global__ __launch_bounds__(256) void pool_cnt_kernel(const int* __restrict__ batch,
                                                       float* __restrict__ cnt, int N) {
  int i = blockIdx.x * blockDim.x + threadIdx.x;
  if (i < N) atomicAdd(&cnt[batch[i]], 1.0f);
}

__global__ __launch_bounds__(256) void pool_sum_kernel(const int* __restrict__ batch,
                                                       const float* __restrict__ h,
                                                       float* __restrict__ pooled,
                                                       int total /* N*256 */) {
  int i = blockIdx.x * blockDim.x + threadIdx.x;
  if (i >= total) return;
  int n = i >> 8, c = i & 255;
  atomicAdd(&pooled[(size_t)batch[n] * 256 + c], h[i]);
}

__global__ __launch_bounds__(256) void pool_div_kernel(const float* __restrict__ cnt,
                                                       float* __restrict__ pooled,
                                                       unsigned short* __restrict__ pooledb,
                                                       int total /* G*256 */) {
  int i = blockIdx.x * blockDim.x + threadIdx.x;
  if (i >= total) return;
  int g = i >> 8;
  float v = pooled[i] / fmaxf(cnt[g], 1.0f);
  pooled[i] = v;
  pooledb[i] = f2bf(v);
}

__global__ __launch_bounds__(256) void fc2_kernel(const float* __restrict__ x /* G x 128 */,
                                                  const float* __restrict__ W /* 128 x 2 */,
                                                  const float* __restrict__ b,
                                                  float* __restrict__ out, int total /* G*2 */) {
  int i = blockIdx.x * blockDim.x + threadIdx.x;
  if (i >= total) return;
  int g = i >> 1, j = i & 1;
  const float* xp = x + (size_t)g * 128;
  float s = b[j];
#pragma unroll 8
  for (int k = 0; k < 128; ++k) s += xp[k] * W[k * 2 + j];
  out[i] = s;
}

// ---------------------------------------------------------------- launcher
extern "C" void kernel_launch(void* const* d_in, const int* in_sizes, int n_in,
                              void* d_out, int out_size, void* d_ws, size_t ws_size,
                              hipStream_t stream) {
  (void)n_in; (void)out_size; (void)ws_size;

  const float* x    = (const float*)d_in[0];
  const int*   ei   = (const int*)d_in[1];
  /* d_in[2] edge_attr unused by reference */
  const int*   batch= (const int*)d_in[3];
  const float* W1   = (const float*)d_in[4];
  const float* as1  = (const float*)d_in[5];
  const float* ad1  = (const float*)d_in[6];
  const float* b1   = (const float*)d_in[7];
  const float* r1W  = (const float*)d_in[8];
  const float* r1b  = (const float*)d_in[9];
  const float* g1   = (const float*)d_in[10];
  const float* bb1  = (const float*)d_in[11];
  const float* W2   = (const float*)d_in[12];
  const float* as2  = (const float*)d_in[13];
  const float* ad2  = (const float*)d_in[14];
  const float* b2   = (const float*)d_in[15];
  const float* r2W  = (const float*)d_in[16];
  const float* r2b  = (const float*)d_in[17];
  const float* g2   = (const float*)d_in[18];
  const float* bb2  = (const float*)d_in[19];
  const float* fc1W = (const float*)d_in[20];
  const float* fc1b = (const float*)d_in[21];
  const float* fc2W = (const float*)d_in[22];
  const float* fc2b = (const float*)d_in[23];
  float* out = (float*)d_out;

  const int N  = in_sizes[0] / 128;   // 50000
  const int E  = in_sizes[1] / 2;     // 800000
  const int G  = 512;
  const int*   src = ei;
  const int*   dst = ei + E;

  char* ws = (char*)d_ws;
  size_t off = 0;
  auto alloc = [&](size_t bytes) -> void* {
    void* p = ws + off;
    off += (bytes + 255) & ~(size_t)255;
    return p;
  };

  unsigned short* xb    = (unsigned short*)alloc((size_t)N * 128 * 2);
  unsigned short* hb    = (unsigned short*)alloc((size_t)N * 256 * 2);
  unsigned short* w1t   = (unsigned short*)alloc(256 * 128 * 2);   // [256,128] = W1^T
  unsigned short* r1t   = (unsigned short*)alloc(256 * 128 * 2);
  unsigned short* w2t   = (unsigned short*)alloc(256 * 256 * 2);
  unsigned short* r2t   = (unsigned short*)alloc(256 * 256 * 2);
  unsigned short* fc1t  = (unsigned short*)alloc(128 * 256 * 2);   // [128,256] = fc1W^T
  float* lin   = (float*)alloc((size_t)N * 256 * 4);
  float* agg   = (float*)alloc((size_t)N * 256 * 4);
  float* es    = (float*)alloc((size_t)N * 4 * 4);
  float* ed    = (float*)alloc((size_t)N * 4 * 4);
  float* emax  = (float*)alloc((size_t)N * 4 * 4);
  float* den   = (float*)alloc((size_t)N * 4 * 4);
  float* ex    = (float*)alloc((size_t)E * 4 * 4);
  float* bnsum = (float*)alloc(256 * 4);
  float* bnsq  = (float*)alloc(256 * 4);
  float* mean  = (float*)alloc(256 * 4);
  float* rstd  = (float*)alloc(256 * 4);
  float* bias1 = (float*)alloc(256 * 4);
  float* bias2 = (float*)alloc(256 * 4);
  float* pooled = (float*)alloc((size_t)G * 256 * 4);
  float* cnt    = (float*)alloc((size_t)G * 4);
  unsigned short* pooledb = (unsigned short*)alloc((size_t)G * 256 * 2);
  float* fc1out = (float*)alloc((size_t)G * 128 * 4);

  auto blocks = [](int n) { return dim3((unsigned)((n + 255) / 256)); };
  const dim3 T(256);

  // --- bf16 conversions: activations row-major, weights transposed to [Nc,K]
  cvt_bf16_kernel<<<blocks(N * 128), T, 0, stream>>>(x, xb, N * 128);
  cvt_bf16_t_kernel<<<blocks(128 * 256), T, 0, stream>>>(W1,  w1t, 128, 256);
  cvt_bf16_t_kernel<<<blocks(128 * 256), T, 0, stream>>>(r1W, r1t, 128, 256);
  cvt_bf16_t_kernel<<<blocks(256 * 256), T, 0, stream>>>(W2,  w2t, 256, 256);
  cvt_bf16_t_kernel<<<blocks(256 * 256), T, 0, stream>>>(r2W, r2t, 256, 256);
  cvt_bf16_t_kernel<<<blocks(256 * 128), T, 0, stream>>>(fc1W, fc1t, 256, 128);
  add_vec_kernel<<<1, T, 0, stream>>>(b1, r1b, bias1, 256);
  add_vec_kernel<<<1, T, 0, stream>>>(b2, r2b, bias2, 256);

  for (int layer = 0; layer < 2; ++layer) {
    const unsigned short* Ain = (layer == 0) ? xb : hb;
    const int K = (layer == 0) ? 128 : 256;
    const unsigned short* Wl = (layer == 0) ? w1t : w2t;
    const unsigned short* Rl = (layer == 0) ? r1t : r2t;
    const float* biasl = (layer == 0) ? bias1 : bias2;
    const float* asl = (layer == 0) ? as1 : as2;
    const float* adl = (layer == 0) ? ad1 : ad2;
    const float* gl  = (layer == 0) ? g1 : g2;
    const float* bl  = (layer == 0) ? bb1 : bb2;

    dim3 gg(256 / BN, (unsigned)((N + BM - 1) / BM));
    // lin = A @ W (no bias; GAT bias folded in post-aggregation)
    gemm_bf16_wmma<<<gg, T, 0, stream>>>(Ain, Wl, nullptr, lin, N, K, 256, 0);
    // agg = A @ resW + (b + res_b)
    gemm_bf16_wmma<<<gg, T, 0, stream>>>(Ain, Rl, biasl, agg, N, K, 256, 0);

    attn_coef_kernel<<<blocks(N * 4), T, 0, stream>>>(lin, asl, adl, es, ed, N * 4);
    fill_f32_kernel<<<blocks(N * 4), T, 0, stream>>>(emax, -1e30f, N * 4);
    fill_f32_kernel<<<blocks(N * 4), T, 0, stream>>>(den, 0.0f, N * 4);
    edge_logit_max_kernel<<<blocks(E * 4), T, 0, stream>>>(src, dst, es, ed, ex, emax, E * 4);
    edge_exp_den_kernel<<<blocks(E * 4), T, 0, stream>>>(dst, emax, ex, den, E * 4);
    edge_scatter_kernel<<<blocks(E * 64), T, 0, stream>>>(src, dst, lin, ex, den, agg, E * 64);

    fill_f32_kernel<<<1, T, 0, stream>>>(bnsum, 0.0f, 256);
    fill_f32_kernel<<<1, T, 0, stream>>>(bnsq, 0.0f, 256);
    bn_stats_kernel<<<dim3(128), T, 0, stream>>>(agg, bnsum, bnsq, N);
    bn_finalize_kernel<<<1, T, 0, stream>>>(bnsum, bnsq, mean, rstd, N);
    if (layer == 0) {
      bn_apply_kernel<<<blocks(N * 256), T, 0, stream>>>(agg, mean, rstd, gl, bl,
                                                         nullptr, hb, N * 256);
    } else {
      bn_apply_kernel<<<blocks(N * 256), T, 0, stream>>>(agg, mean, rstd, gl, bl,
                                                         lin, nullptr, N * 256);
    }
  }

  // --- mean pooling per graph
  fill_f32_kernel<<<blocks(G), T, 0, stream>>>(cnt, 0.0f, G);
  fill_f32_kernel<<<blocks(G * 256), T, 0, stream>>>(pooled, 0.0f, G * 256);
  pool_cnt_kernel<<<blocks(N), T, 0, stream>>>(batch, cnt, N);
  pool_sum_kernel<<<blocks(N * 256), T, 0, stream>>>(batch, lin, pooled, N * 256);
  pool_div_kernel<<<blocks(G * 256), T, 0, stream>>>(cnt, pooled, pooledb, G * 256);

  // --- MLP head: fc1 via WMMA (bias+relu fused), fc2 tiny
  {
    dim3 gg(128 / BN, (unsigned)((G + BM - 1) / BM));
    gemm_bf16_wmma<<<gg, T, 0, stream>>>(pooledb, fc1t, fc1b, fc1out, G, 256, 128, 1);
  }
  fc2_kernel<<<blocks(G * 2), T, 0, stream>>>(fc1out, fc2W, fc2b, out, G * 2);
}
